// CausalSelfAttention_21131239096808
// MI455X (gfx1250) — matrix operand order repaired and checked by reading, not compiled, not run on hardware
//
#include <hip/hip_runtime.h>
#include <hip/hip_bf16.h>

// ---------------- problem constants ----------------
#define B_  4
#define T_  2048
#define E_  1024
#define H_  16
#define D_  64
#define BT  (B_ * T_)          // 8192 rows

typedef __bf16 bf16_t;
typedef bf16_t v16bf __attribute__((ext_vector_type(16)));
typedef bf16_t v8bf  __attribute__((ext_vector_type(8)));
typedef float  v8f   __attribute__((ext_vector_type(8)));

#if defined(__has_builtin)
#  if __has_builtin(__builtin_amdgcn_tensor_load_to_lds)
#    define HAVE_TDM 1
#  endif
#endif

// ---------------- bf16 helpers (RNE) ----------------
__device__ __forceinline__ bf16_t f2bf(float f) {
    union { float f; unsigned u; } v; v.f = f;
    unsigned r = v.u + 0x7FFFu + ((v.u >> 16) & 1u);
    union { unsigned short s; bf16_t b; } o; o.s = (unsigned short)(r >> 16);
    return o.b;
}
__device__ __forceinline__ float bf2f(bf16_t b) {
    union { bf16_t b; unsigned short s; } i; i.b = b;
    union { unsigned u; float f; } v; v.u = ((unsigned)i.s) << 16;
    return v.f;
}

// A-operand (16x32, 16-bit): lane m = L&15; elems 0..7 = K kb..kb+7, elems 8..15 = K 16+kb..+7,
// kb = (L>>4)*8.  (ISA 7.12.2, 16-bit A table)
__device__ __forceinline__ v16bf loadA16(const bf16_t* row, int kb) {
    v8bf lo = *(const v8bf*)(row + kb);
    v8bf hi = *(const v8bf*)(row + kb + 16);
    return __builtin_shufflevector(lo, hi, 0,1,2,3,4,5,6,7,8,9,10,11,12,13,14,15);
}
// B-operand (32x16, 16-bit): lane n = L&15; elems 0..7 = K kb..kb+7, elems 8..15 = K 16+kb..+7
// (caller adds kb = 8*(L>>4)), matching the A table order.
__device__ __forceinline__ v16bf loadB16(const bf16_t* p) {
    v8bf lo = *(const v8bf*)(p);
    v8bf hi = *(const v8bf*)(p + 16);
    return __builtin_shufflevector(lo, hi, 0,1,2,3,4,5,6,7,8,9,10,11,12,13,14,15);
}

// async 16B global -> LDS copy (per lane), tracked by ASYNCcnt (fallback staging path)
__device__ __forceinline__ void async_g2l_b128(unsigned lds_byte_off, const void* gaddr) {
    asm volatile("global_load_async_to_lds_b128 %0, %1, off"
                 :: "v"(lds_byte_off), "v"((unsigned long long)gaddr) : "memory");
}

#if HAVE_TDM
// TDM 2D tile load: build D# groups 0/1 per ISA 8.3/8.4 and issue TENSOR_LOAD_TO_LDS.
// data_size = 2 bytes (bf16). Tile is tile1 rows x tile0 contiguous elements,
// row stride = stride0 elements; packed contiguously into LDS at ldsOff.
__device__ __forceinline__ void tdm_load_2d(unsigned ldsOff, const void* gaddr,
                                            unsigned td0, unsigned td1,
                                            unsigned tile0, unsigned tile1,
                                            unsigned long long stride0) {
    typedef unsigned v4u __attribute__((ext_vector_type(4)));
    typedef int      v4i __attribute__((ext_vector_type(4)));
    typedef int      v8i __attribute__((ext_vector_type(8)));
    const unsigned long long ga = (unsigned long long)gaddr;
    v4u g0;
    g0[0] = 1u;                                             // count=1, user mode
    g0[1] = ldsOff;                                         // lds_addr (bytes)
    g0[2] = (unsigned)ga;                                   // global_addr[31:0]
    g0[3] = (unsigned)((ga >> 32) & 0x1FFFFFFu) | (2u << 30); // addr[56:32] | type=2
    v8i g1;
    g1[0] = (int)(1u << 16);                                // data_size=1 (2 bytes)
    g1[1] = (int)((td0 & 0xFFFFu) << 16);                   // tensor_dim0[15:0] @ bits 63:48
    g1[2] = (int)(((td0 >> 16) & 0xFFFFu) | ((td1 & 0xFFFFu) << 16)); // dim0 hi | dim1 lo
    g1[3] = (int)(((td1 >> 16) & 0xFFFFu) | (tile0 << 16)); // dim1 hi | tile_dim0
    g1[4] = (int)(tile1 & 0xFFFFu);                         // tile_dim1 | tile_dim2=0
    g1[5] = (int)(unsigned)(stride0 & 0xFFFFFFFFull);       // tensor_dim0_stride[31:0]
    g1[6] = (int)(unsigned)((stride0 >> 32) & 0xFFFFull);   // stride0[47:32] | stride1 lo=0
    g1[7] = 0;                                              // tensor_dim1_stride hi = 0
    v4i z4 = {0, 0, 0, 0};
#if __clang_major__ >= 23
    v8i z8 = {0, 0, 0, 0, 0, 0, 0, 0};
    __builtin_amdgcn_tensor_load_to_lds(g0, g1, z4, z4, z8, 0);
#else
    __builtin_amdgcn_tensor_load_to_lds(g0, g1, z4, z4, 0);
#endif
}
#endif

// ---------------- kernel 1: f32 -> bf16 ----------------
__global__ void cvt_f32_bf16(const float* __restrict__ src, bf16_t* __restrict__ dst, int n) {
    int i = blockIdx.x * blockDim.x + threadIdx.x;
    if (i < n) dst[i] = f2bf(src[i]);
}

// ---------------- kernel 2: fused QKV projection, 64x64 register-blocked ----------------
__global__ void __launch_bounds__(256)
qkv_proj(const bf16_t* __restrict__ X,
         const bf16_t* __restrict__ Wq, const bf16_t* __restrict__ Wk, const bf16_t* __restrict__ Wv,
         const float*  __restrict__ bq, const float*  __restrict__ bk, const float*  __restrict__ bv,
         bf16_t* __restrict__ Qb, bf16_t* __restrict__ Kb, bf16_t* __restrict__ Vt)
{
    const int lane = threadIdx.x & 31;
    const int wave = blockIdx.x * (blockDim.x >> 5) + (threadIdx.x >> 5);
    const int tilesPerProj = (BT / 64) * (E_ / 64);      // 128 * 16 = 2048
    const int proj = wave / tilesPerProj;
    if (proj >= 3) return;
    const int rem = wave - proj * tilesPerProj;
    const int mt = rem / (E_ / 64);
    const int nt = rem % (E_ / 64);
    const int m0 = mt * 64, n0 = nt * 64;

    const int l15  = lane & 15;
    const int kb8  = (lane >> 4) * 8;
    const int kh16 = (lane >> 4) * 16;

    const bf16_t* W    = (proj == 0) ? Wq : (proj == 1) ? Wk : Wv;
    const float*  bias = (proj == 0) ? bq : (proj == 1) ? bk : bv;

    const bf16_t* arow[4];
    const bf16_t* brow[4];
    #pragma unroll
    for (int i = 0; i < 4; ++i) {
        arow[i] = X + (size_t)(m0 + i * 16 + l15) * E_;
        brow[i] = W + (size_t)(n0 + i * 16 + l15) * E_ + kb8;
    }

    v8f acc[4][4] = {};
    for (int k0 = 0; k0 < E_; k0 += 32) {
        v16bf A[4], Bv[4];
        #pragma unroll
        for (int mi = 0; mi < 4; ++mi) A[mi] = loadA16(arow[mi] + k0, kb8);
        #pragma unroll
        for (int ni = 0; ni < 4; ++ni) Bv[ni] = loadB16(brow[ni] + k0);
        #pragma unroll
        for (int mi = 0; mi < 4; ++mi)
            #pragma unroll
            for (int ni = 0; ni < 4; ++ni)
                acc[mi][ni] = __builtin_amdgcn_wmma_f32_16x16x32_bf16(
                    false, A[mi], false, Bv[ni], (short)0, acc[mi][ni], false, false);
    }

    #pragma unroll
    for (int ni = 0; ni < 4; ++ni) {
        const int n = n0 + ni * 16 + l15;
        const float bb = bias[n];
        const int h = n >> 6, dch = n & 63;
        #pragma unroll
        for (int mi = 0; mi < 4; ++mi) {
            #pragma unroll
            for (int r = 0; r < 8; ++r) {
                const int mg   = m0 + mi * 16 + r + kb8;
                const int bidx = mg >> 11;
                const int t    = mg & (T_ - 1);
                const bf16_t val = f2bf(acc[mi][ni][r] + bb);
                if (proj == 0)      Qb[(((size_t)(bidx * H_ + h) * T_) + t) * D_ + dch] = val;
                else if (proj == 1) Kb[(((size_t)(bidx * H_ + h) * T_) + t) * D_ + dch] = val;
                else                Vt[(((size_t)(bidx * H_ + h) * D_) + dch) * T_ + t] = val;
            }
        }
    }
}

// ---------------- kernel 3: RoPE on Q and K (in place, pairwise) ----------------
__global__ void rope_qk(bf16_t* __restrict__ Qb, bf16_t* __restrict__ Kb) {
    const int i = blockIdx.x * blockDim.x + threadIdx.x;    // over B*H*T*32
    if (i >= B_ * H_ * T_ * (D_ / 2)) return;
    const int j  = i & 31;
    const int t  = (i >> 5) & (T_ - 1);
    const int bh = i >> 16;
    const float theta = __expf(-(float)j * (1.0f / 32.0f) * 9.210340372f); // ln(10000)
    const float ang = (float)(t + 1) * theta;                // positions start at 1
    const float cs = __cosf(ang), sn = __sinf(ang);
    const size_t base = ((size_t)bh * T_ + t) * D_;
    {
        float y0 = bf2f(Qb[base + j]), y1 = bf2f(Qb[base + j + 32]);
        Qb[base + j]      = f2bf(y0 * cs - y1 * sn);
        Qb[base + j + 32] = f2bf(y1 * cs + y0 * sn);
    }
    {
        float y0 = bf2f(Kb[base + j]), y1 = bf2f(Kb[base + j + 32]);
        Kb[base + j]      = f2bf(y0 * cs - y1 * sn);
        Kb[base + j + 32] = f2bf(y1 * cs + y0 * sn);
    }
}

// ---------------- kernel 4: block-cooperative causal flash attention ----------------
// Block = 8 waves sharing one (b,h) and 128 consecutive queries. K/V tiles (32 keys)
// are staged ONCE per block into LDS: via the Tensor Data Mover (one 2D descriptor per
// tile, TENSORcnt) when available, else per-lane global_load_async_to_lds_b128
// (ASYNCcnt). Block-uniform key loop => __syncthreads is safe; fully-masked tiles
// give p=0 / alpha=1 so the online softmax is unchanged.
__global__ void __launch_bounds__(256)
attn_fwd(const bf16_t* __restrict__ Qb, const bf16_t* __restrict__ Kb,
         const bf16_t* __restrict__ Vt, bf16_t* __restrict__ Ob)
{
    __shared__ __align__(16) unsigned char smraw[4096 + 4096 + 8 * 512 * 4];
    bf16_t* Ksm = (bf16_t*)(smraw);            // [32 keys][64 d]   4 KB
    bf16_t* Vsm = (bf16_t*)(smraw + 4096);     // [64 d][32 keys]   4 KB
    float*  Psm = (float*)(smraw + 8192);      // per-wave 16x32 P staging, 16 KB

    const int tid  = threadIdx.x;
    const int lane = tid & 31;
    const int wIn  = tid >> 5;
    const int qblk = blockIdx.x & 15;          // 16 query blocks of 128
    const int bh   = blockIdx.x >> 4;          // 0..63 (b*16 + h)
    const int qb0  = qblk * 128;
    const int q0   = qb0 + wIn * 16;           // this wave's 16 queries
    const int l15  = lane & 15;
    const int kb8  = (lane >> 4) * 8;
    const int kh16 = (lane >> 4) * 16;
    float* myP = Psm + wIn * 512;

    const bf16_t* Kbase = Kb + (size_t)bh * T_ * D_;
    const bf16_t* Vbase = Vt + (size_t)bh * D_ * T_;

    // staging assignment (fallback path): 256 threads x 16B cover each 4KB tile
    const int krow_s = tid >> 3;               // 0..31 (key)
    const int kcol_s = (tid & 7) * 8;          // d offset (bf16 elems)
    const int vrow_s = tid >> 2;               // 0..63 (d)
    const int vcol_s = (tid & 3) * 8;          // key offset (bf16 elems)
    const unsigned ldsK = (unsigned)(unsigned long long)(Ksm + krow_s * 64 + kcol_s);
    const unsigned ldsV = (unsigned)(unsigned long long)(Vsm + vrow_s * 32 + vcol_s);

    // Q tile (16x64) as two A operands (d in [0,32) and [32,64))
    const bf16_t* qrow = Qb + ((size_t)bh * T_ + q0 + l15) * D_;
    v16bf aQ[2];
    #pragma unroll
    for (int w = 0; w < 2; ++w) aQ[w] = loadA16(qrow + w * 32, kb8);

    v8f oacc[4] = {};
    float mrow[8], lrow[8];
    #pragma unroll
    for (int r = 0; r < 8; ++r) { mrow[r] = -3.0e38f; lrow[r] = 0.0f; }
    const float scale = 0.125f;                // 1/sqrt(64)

    for (int kb0 = 0; kb0 <= qb0 + 127; kb0 += 32) {
#if HAVE_TDM
        // ---- TDM staging: wave0 pulls the K tile, wave1 the V tile
        if (wIn == 0) {
            tdm_load_2d((unsigned)(unsigned long long)Ksm, Kbase + (size_t)kb0 * D_,
                        /*td0=*/64, /*td1=*/2048, /*tile0=*/64, /*tile1=*/32, /*stride0=*/64);
            __builtin_amdgcn_s_wait_tensorcnt(0);
        } else if (wIn == 1) {
            tdm_load_2d((unsigned)(unsigned long long)Vsm, Vbase + kb0,
                        /*td0=*/2048, /*td1=*/64, /*tile0=*/32, /*tile1=*/64, /*stride0=*/2048);
            __builtin_amdgcn_s_wait_tensorcnt(0);
        }
#else
        // ---- async per-lane staging (ASYNCcnt)
        const bf16_t* gK = Kbase + (size_t)(kb0 + krow_s) * D_ + kcol_s;
        const bf16_t* gV = Vbase + (size_t)vrow_s * T_ + kb0 + vcol_s;
        async_g2l_b128(ldsK, gK);
        async_g2l_b128(ldsV, gV);
        asm volatile("s_wait_asynccnt 0x0" ::: "memory");
#endif
        if (kb0 + 32 <= qb0 + 127)   // prefetch next key tile (global_prefetch_b8)
            __builtin_prefetch(Kbase + (size_t)(kb0 + 32 + krow_s) * D_ + kcol_s, 0, 0);
        __syncthreads();

        // ---- S = Q K^T (16 queries x 32 keys) from LDS-resident K
        v8f sAcc[2] = {};
        #pragma unroll
        for (int hN = 0; hN < 2; ++hN) {
            const bf16_t* krow = Ksm + (hN * 16 + l15) * 64;
            #pragma unroll
            for (int w = 0; w < 2; ++w) {
                v16bf bK = loadB16(krow + w * 32 + kb8);
                sAcc[hN] = __builtin_amdgcn_wmma_f32_16x16x32_bf16(
                    false, aQ[w], false, bK, (short)0, sAcc[hN], false, false);
            }
        }

        // ---- online softmax (rows live in one VGPR element across a 16-lane half)
        #pragma unroll
        for (int r = 0; r < 8; ++r) {
            const int qg  = q0 + r + kb8;
            const int k0g = kb0 + l15;
            const int k1g = kb0 + 16 + l15;
            float s0 = sAcc[0][r] * scale; if (k0g > qg) s0 = -3.0e38f;  // causal mask
            float s1 = sAcc[1][r] * scale; if (k1g > qg) s1 = -3.0e38f;
            float mx = fmaxf(s0, s1);
            #pragma unroll
            for (int msk = 1; msk < 16; msk <<= 1) mx = fmaxf(mx, __shfl_xor(mx, msk, 32));
            const float mnew  = fmaxf(mrow[r], mx);
            const float alpha = __expf(mrow[r] - mnew);
            const float p0 = __expf(s0 - mnew);
            const float p1 = __expf(s1 - mnew);
            float ps = p0 + p1;
            #pragma unroll
            for (int msk = 1; msk < 16; msk <<= 1) ps += __shfl_xor(ps, msk, 32);
            lrow[r] = lrow[r] * alpha + ps;
            mrow[r] = mnew;
            #pragma unroll
            for (int j = 0; j < 4; ++j) oacc[j][r] *= alpha;
            const int M = r + kb8;                       // stage P (C layout) into LDS
            myP[M * 32 + l15]      = p0;
            myP[M * 32 + 16 + l15] = p1;
        }
        asm volatile("s_wait_dscnt 0x0" ::: "memory");   // wave-local P RAW

        // ---- rebuild P in A-operand layout, convert to bf16
        v16bf pA;
        #pragma unroll
        for (int i = 0; i < 8; ++i) {
            pA[i]     = f2bf(myP[l15 * 32 + kb8 + i]);
            pA[i + 8] = f2bf(myP[l15 * 32 + 16 + kb8 + i]);
        }

        // ---- O += P * V from LDS-resident transposed V
        #pragma unroll
        for (int j = 0; j < 4; ++j) {
            v16bf bV = loadB16(Vsm + (j * 16 + l15) * 32 + kb8);
            oacc[j] = __builtin_amdgcn_wmma_f32_16x16x32_bf16(
                false, pA, false, bV, (short)0, oacc[j], false, false);
        }
        __syncthreads();                                 // WAR: all waves done with K/V tile
    }

    // ---- normalize and store O as bf16 in [b, t, h*64+d] layout
    const int bidx = bh >> 4, h = bh & 15;
    #pragma unroll
    for (int r = 0; r < 8; ++r) {
        const float inv = 1.0f / lrow[r];
        const int mg = q0 + r + kb8;
        const size_t orow = ((size_t)bidx * T_ + mg) * E_ + h * D_;
        #pragma unroll
        for (int j = 0; j < 4; ++j)
            Ob[orow + j * 16 + l15] = f2bf(oacc[j][r] * inv);
    }
}

// ---------------- kernel 5: output projection, 64x64 register-blocked ----------------
__global__ void __launch_bounds__(256)
out_proj(const bf16_t* __restrict__ Ob, const bf16_t* __restrict__ Wo,
         const float* __restrict__ bo, float* __restrict__ out)
{
    const int lane = threadIdx.x & 31;
    const int wave = blockIdx.x * (blockDim.x >> 5) + (threadIdx.x >> 5);
    const int mt = wave / (E_ / 64);
    const int nt = wave % (E_ / 64);
    if (mt >= BT / 64) return;
    const int m0 = mt * 64, n0 = nt * 64;
    const int l15  = lane & 15;
    const int kb8  = (lane >> 4) * 8;
    const int kh16 = (lane >> 4) * 16;

    const bf16_t* arow[4];
    const bf16_t* brow[4];
    #pragma unroll
    for (int i = 0; i < 4; ++i) {
        arow[i] = Ob + (size_t)(m0 + i * 16 + l15) * E_;
        brow[i] = Wo + (size_t)(n0 + i * 16 + l15) * E_ + kb8;
    }

    v8f acc[4][4] = {};
    for (int k0 = 0; k0 < E_; k0 += 32) {
        v16bf A[4], Bv[4];
        #pragma unroll
        for (int mi = 0; mi < 4; ++mi) A[mi] = loadA16(arow[mi] + k0, kb8);
        #pragma unroll
        for (int ni = 0; ni < 4; ++ni) Bv[ni] = loadB16(brow[ni] + k0);
        #pragma unroll
        for (int mi = 0; mi < 4; ++mi)
            #pragma unroll
            for (int ni = 0; ni < 4; ++ni)
                acc[mi][ni] = __builtin_amdgcn_wmma_f32_16x16x32_bf16(
                    false, A[mi], false, Bv[ni], (short)0, acc[mi][ni], false, false);
    }

    #pragma unroll
    for (int ni = 0; ni < 4; ++ni) {
        const int n = n0 + ni * 16 + l15;
        const float bb = bo[n];
        #pragma unroll
        for (int mi = 0; mi < 4; ++mi) {
            #pragma unroll
            for (int r = 0; r < 8; ++r) {
                const int mg = m0 + mi * 16 + r + kb8;
                out[(size_t)mg * E_ + n] = acc[mi][ni][r] + bb;
            }
        }
    }
}

// ---------------- host launcher ----------------
extern "C" void kernel_launch(void* const* d_in, const int* in_sizes, int n_in,
                              void* d_out, int out_size, void* d_ws, size_t ws_size,
                              hipStream_t stream) {
    const float* x  = (const float*)d_in[0];
    const float* Wq = (const float*)d_in[1];
    const float* bq = (const float*)d_in[2];
    const float* Wk = (const float*)d_in[3];
    const float* bk = (const float*)d_in[4];
    const float* Wv = (const float*)d_in[5];
    const float* bv = (const float*)d_in[6];
    const float* Wo = (const float*)d_in[7];
    const float* bo = (const float*)d_in[8];
    float* out = (float*)d_out;

    char* ws = (char*)d_ws;
    bf16_t* xb  = (bf16_t*)(ws + 0);          // 16 MB : x in bf16
    bf16_t* wqb = (bf16_t*)(ws + 16777216);   //  2 MB
    bf16_t* wkb = (bf16_t*)(ws + 18874368);   //  2 MB
    bf16_t* wvb = (bf16_t*)(ws + 20971520);   //  2 MB
    bf16_t* wob = (bf16_t*)(ws + 23068672);   //  2 MB
    bf16_t* Qb  = (bf16_t*)(ws + 25165824);   // 16 MB : [b,h,t,d]
    bf16_t* Kb  = (bf16_t*)(ws + 41943040);   // 16 MB : [b,h,t,d]
    bf16_t* Vt  = (bf16_t*)(ws + 58720256);   // 16 MB : [b,h,d,t] (transposed)
    bf16_t* Obf = (bf16_t*)(ws + 75497472);   // 16 MB : [b,t,e]
    // total 92274688 bytes

    cvt_f32_bf16<<<(BT * E_ + 255) / 256, 256, 0, stream>>>(x,  xb,  BT * E_);
    cvt_f32_bf16<<<(E_ * E_ + 255) / 256, 256, 0, stream>>>(Wq, wqb, E_ * E_);
    cvt_f32_bf16<<<(E_ * E_ + 255) / 256, 256, 0, stream>>>(Wk, wkb, E_ * E_);
    cvt_f32_bf16<<<(E_ * E_ + 255) / 256, 256, 0, stream>>>(Wv, wvb, E_ * E_);
    cvt_f32_bf16<<<(E_ * E_ + 255) / 256, 256, 0, stream>>>(Wo, wob, E_ * E_);

    // 3 projections * (128*16) 64x64 tiles = 6144 waves / 8 per block
    qkv_proj<<<6144 / 8, 256, 0, stream>>>(xb, wqb, wkb, wvb, bq, bk, bv, Qb, Kb, Vt);

    rope_qk<<<(B_ * H_ * T_ * 32 + 255) / 256, 256, 0, stream>>>(Qb, Kb);

    // 64 (b,h) x 16 query-blocks = 1024 blocks of 8 cooperating waves
    attn_fwd<<<1024, 256, 0, stream>>>(Qb, Kb, Vt, Obf);

    // (128*16) 64x64 tiles = 2048 waves / 8 per block
    out_proj<<<2048 / 8, 256, 0, stream>>>(Obf, wob, bo, out);
}
